// AttentionLayer_57561151701380
// MI455X (gfx1250) — matrix operand order repaired and checked by reading, not compiled, not run on hardware
//
#include <hip/hip_runtime.h>
#include <hip/hip_bf16.h>

#define SEQ  8192
#define DIMK 1024

typedef __attribute__((ext_vector_type(16))) __bf16        bf16x16;
typedef __attribute__((ext_vector_type(8)))  float         f32x8;
typedef __attribute__((ext_vector_type(8)))  unsigned short u16x8;

union FragU {
    bf16x16        v;
    u16x8          h[2];
    __bf16         e[16];
};

union HalfFrag {
    u16x8  h;
    __bf16 e[8];
};

__device__ __forceinline__ f32x8 wmma_bf16(const FragU& a, const FragU& b, f32x8 c) {
    return __builtin_amdgcn_wmma_f32_16x16x32_bf16(
        /*neg_a=*/false, a.v, /*neg_b=*/false, b.v,
        /*c_mod=*/(short)0, c, /*reuse_a=*/false, /*reuse_b=*/false);
}

// ---------------------------------------------------------------------------
// Projection GEMM: C[bf16] = (A[f32, MxK] @ B[f32, KxN]) * scale
// transposeC=1 stores C transposed as C[n*SEQ + m] (used for V); in that mode
// the 8 rows per accumulator are contiguous -> packed global_store_b128.
// Block = 256 threads = 8 waves; block tile 128x128, wave tile 32x64.
// ---------------------------------------------------------------------------
__global__ __launch_bounds__(256) void proj_gemm(const float* __restrict__ A,
                                                 const float* __restrict__ B,
                                                 __bf16* __restrict__ C,
                                                 float scale, int transposeC) {
    const int K = DIMK, N = DIMK;
    const int tid = threadIdx.x;
    const int w = tid >> 5, l = tid & 31;
    const int l16 = l & 15, lh = l >> 4;
    const int Mbase = blockIdx.x * 128 + (w & 3) * 32;
    const int Nbase = blockIdx.y * 128 + (w >> 2) * 64;

    const f32x8 z8 = {0.f,0.f,0.f,0.f,0.f,0.f,0.f,0.f};
    f32x8 acc[2][4];
#pragma unroll
    for (int t = 0; t < 2; ++t)
#pragma unroll
        for (int c = 0; c < 4; ++c) acc[t][c] = z8;

    for (int kk = 0; kk < K; kk += 32) {
        // A fragments: lane holds row m, K chunks [lh*8 .. +7] and [lh*8+16 .. +23]
        FragU a[2];
#pragma unroll
        for (int t = 0; t < 2; ++t) {
            const float* ap = A + (size_t)(Mbase + t * 16 + l16) * K + kk + lh * 8;
#pragma unroll
            for (int e = 0; e < 8; ++e) {
                a[t].e[e]     = (__bf16)ap[e];        // native v_cvt_pk_bf16_f32
                a[t].e[8 + e] = (__bf16)ap[16 + e];
            }
        }
        // B fragments: lane holds column n, k = kk + lh*8 + (e&7) + (e>>3)*16 (strided f32 reads,
        // coalesced across lanes at each e)
#pragma unroll
        for (int c = 0; c < 4; ++c) {
            FragU b;
            const float* bp = B + (size_t)(kk + lh * 8) * N + Nbase + c * 16 + l16;
#pragma unroll
            for (int e = 0; e < 16; ++e) b.e[e] = (__bf16)bp[(size_t)((e & 7) + (e >> 3) * 16) * N];
#pragma unroll
            for (int t = 0; t < 2; ++t) acc[t][c] = wmma_bf16(a[t], b, acc[t][c]);
        }
    }

    if (transposeC) {
#pragma unroll
        for (int t = 0; t < 2; ++t)
#pragma unroll
            for (int c = 0; c < 4; ++c) {
                HalfFrag st;
#pragma unroll
                for (int r = 0; r < 8; ++r) st.e[r] = (__bf16)(acc[t][c][r] * scale);
                const int n = Nbase + c * 16 + l16;
                const int m0 = Mbase + t * 16 + lh * 8;     // 8 consecutive rows
                *(u16x8*)(C + (size_t)n * SEQ + m0) = st.h; // global_store_b128
            }
    } else {
#pragma unroll
        for (int t = 0; t < 2; ++t)
#pragma unroll
            for (int c = 0; c < 4; ++c)
#pragma unroll
                for (int r = 0; r < 8; ++r) {
                    const int m = Mbase + t * 16 + r + lh * 8;
                    const int n = Nbase + c * 16 + l16;
                    C[(size_t)m * N + n] = (__bf16)(acc[t][c][r] * scale);
                }
    }
}

// ---------------------------------------------------------------------------
// Flash attention: one block = 32 query rows, 256 threads = 8 waves.
// Qp/Kp row-major bf16 [SEQ][D]; Vt transposed bf16 [D][SEQ]; Out f32.
// Qp is pre-scaled by 1/sqrt(D).
// ---------------------------------------------------------------------------
__global__ __launch_bounds__(256) void flash_attn(const __bf16* __restrict__ Qp,
                                                  const __bf16* __restrict__ Kp,
                                                  const __bf16* __restrict__ Vt,
                                                  float* __restrict__ Out) {
    const int D = DIMK;
    __shared__ float  ST[32 * 32];   // S tile, [query i][key j]
    __shared__ __bf16 Pm[32 * 32];   // P tile bf16, [i][j]
    __shared__ float mArr[32], lArr[32], aArr[32];

    const int tid = threadIdx.x;
    const int w = tid >> 5, l = tid & 31;
    const int l16 = l & 15, lh = l >> 4;
    const int qb = blockIdx.x * 32;

    // Phase-A role: (q-tile, k-tile, d-half); Phase-B column strip
    const int qt = w & 1, kt = (w >> 1) & 1, dh = (w >> 2) & 1;
    const int wcol = w * 128;

    const f32x8 z8 = {0.f,0.f,0.f,0.f,0.f,0.f,0.f,0.f};
    f32x8 acc[2][8];
#pragma unroll
    for (int t = 0; t < 2; ++t)
#pragma unroll
        for (int c = 0; c < 8; ++c) acc[t][c] = z8;

    if (tid < 32) { mArr[tid] = -3.0e38f; lArr[tid] = 0.f; }
    __syncthreads();

    const __bf16* qrow = Qp + (size_t)(qb + qt * 16 + l16) * D;
    const int nkb = qb / 32 + 1;

    for (int ib = 0; ib < nkb; ++ib) {
        const int kb = ib * 32;

        // zero S tile (1024 floats / 256 threads)
#pragma unroll
        for (int e = 0; e < 4; ++e) ST[tid * 4 + e] = 0.f;

        // prefetch next key block rows + next V columns (global_prefetch_b8)
        if (ib + 1 < nkb) {
            __builtin_prefetch(Kp + (size_t)(kb + 32 + kt * 16 + l16) * D + dh * 512, 0, 1);
            __builtin_prefetch(Vt + (size_t)(wcol + l16) * SEQ + kb + 32, 0, 1);
        }
        __syncthreads();

        // ---- Phase A: partial S = Q @ K^T over this wave's d-half ----
        f32x8 sacc = z8;
        const __bf16* krow = Kp + (size_t)(kb + kt * 16 + l16) * D;
#pragma unroll 4
        for (int s = 0; s < 16; ++s) {
            const int dd = dh * 512 + s * 32;
            FragU a, b;
            a.h[0] = *(const u16x8*)(qrow + dd + lh * 8);        // K = dd+lh*8 .. +7
            a.h[1] = *(const u16x8*)(qrow + dd + lh * 8 + 16);   // K = +16 .. +23
            b.h[0] = *(const u16x8*)(krow + dd + lh * 8);        // col j, K = dd+lh*8 .. +7
            b.h[1] = *(const u16x8*)(krow + dd + lh * 8 + 16);
            sacc = wmma_bf16(a, b, sacc);
        }
#pragma unroll
        for (int r = 0; r < 8; ++r) {
            const int i = qt * 16 + r + lh * 8;
            const int j = kt * 16 + l16;
            atomicAdd(&ST[i * 32 + j], sacc[r]);
        }
        __syncthreads();

        // ---- Online softmax (wave 0, one query row per lane) ----
        if (w == 0) {
            const int i = l;
            const int jlim = (kb + 31 <= qb + i) ? 31 : (qb + i - kb); // causal
            const float m_old = mArr[i];
            float mx = m_old;
            for (int j = 0; j <= jlim; ++j) mx = fmaxf(mx, ST[i * 32 + j]);
            const float alpha = __expf(m_old - mx);
            float ls = 0.f;
            __bf16 prow[32];
#pragma unroll 8
            for (int j = 0; j < 32; ++j) {
                const float p = (j <= jlim) ? __expf(ST[i * 32 + j] - mx) : 0.f;
                prow[j] = (__bf16)p;
                ls += p;
            }
#pragma unroll
            for (int g = 0; g < 4; ++g)                      // 4x ds_store_b128
                *(u16x8*)&Pm[i * 32 + g * 8] = *(const u16x8*)&prow[g * 8];
            lArr[i] = lArr[i] * alpha + ls;
            mArr[i] = mx;
            aArr[i] = alpha;
        }
        __syncthreads();

        // ---- Phase B: O = diag(alpha)*O + P @ V (this wave's 128 columns) ----
        float as[2][8];
#pragma unroll
        for (int t = 0; t < 2; ++t)
#pragma unroll
            for (int r = 0; r < 8; ++r) as[t][r] = aArr[t * 16 + r + lh * 8];

        FragU pA[2];
#pragma unroll
        for (int t = 0; t < 2; ++t) {
            const __bf16* pp = &Pm[(t * 16 + l16) * 32 + lh * 8];
            pA[t].h[0] = *(const u16x8*)pp;
            pA[t].h[1] = *(const u16x8*)(pp + 16);
        }
#pragma unroll
        for (int c = 0; c < 8; ++c) {
            FragU b;  // column n of V block: contiguous keys from transposed Vt
            const __bf16* vp =
                Vt + (size_t)(wcol + c * 16 + l16) * SEQ + kb + lh * 8;
            b.h[0] = *(const u16x8*)(vp);
            b.h[1] = *(const u16x8*)(vp + 16);
#pragma unroll
            for (int t = 0; t < 2; ++t) {
                f32x8 x = acc[t][c];
#pragma unroll
                for (int r = 0; r < 8; ++r) x[r] *= as[t][r];
                acc[t][c] = wmma_bf16(pA[t], b, x);
            }
        }
        __syncthreads();
    }

    // ---- Epilogue: divide by softmax denominator, store f32 ----
    float linv[2][8];
#pragma unroll
    for (int t = 0; t < 2; ++t)
#pragma unroll
        for (int r = 0; r < 8; ++r) linv[t][r] = 1.0f / lArr[t * 16 + r + lh * 8];
#pragma unroll
    for (int t = 0; t < 2; ++t)
#pragma unroll
        for (int c = 0; c < 8; ++c)
#pragma unroll
            for (int r = 0; r < 8; ++r) {
                const int row = qb + t * 16 + r + lh * 8;
                const int col = wcol + c * 16 + l16;
                Out[(size_t)row * DIMK + col] = acc[t][c][r] * linv[t][r];
            }
}

// ---------------------------------------------------------------------------
extern "C" void kernel_launch(void* const* d_in, const int* in_sizes, int n_in,
                              void* d_out, int out_size, void* d_ws, size_t ws_size,
                              hipStream_t stream) {
    const float* q  = (const float*)d_in[0];
    const float* k  = (const float*)d_in[1];
    const float* v  = (const float*)d_in[2];
    const float* Wq = (const float*)d_in[3];
    const float* Wk = (const float*)d_in[4];
    const float* Wv = (const float*)d_in[5];
    float* out = (float*)d_out;

    __bf16* Qp = (__bf16*)d_ws;                        // [SEQ][D] bf16
    __bf16* Kp = Qp + (size_t)SEQ * DIMK;              // [SEQ][D] bf16
    __bf16* Vt = Kp + (size_t)SEQ * DIMK;              // [D][SEQ] bf16 (transposed)

    dim3 pgrid(SEQ / 128, DIMK / 128);
    // fold 1/sqrt(D_K)=1/32 into Q projection (mask -> -inf is scale-invariant)
    proj_gemm<<<pgrid, 256, 0, stream>>>(q, Wq, Qp, 0.03125f, 0);
    proj_gemm<<<pgrid, 256, 0, stream>>>(k, Wk, Kp, 1.0f,     0);
    proj_gemm<<<pgrid, 256, 0, stream>>>(v, Wv, Vt, 1.0f,     1);

    flash_attn<<<SEQ / 32, 256, 0, stream>>>(Qp, Kp, Vt, out);
}
